// AttentionModule_6794638262728
// MI455X (gfx1250) — matrix operand order repaired and checked by reading, not compiled, not run on hardware
//
#include <hip/hip_runtime.h>
#include <hip/hip_bf16.h>

// ---------------------------------------------------------------------------
// Problem constants (from reference): B=64, N=512, E=D=128, H=4
// ---------------------------------------------------------------------------
constexpr int kB = 64;
constexpr int kN = 512;
constexpr int kE = 128;
constexpr int kD = 128;
constexpr int kH = 4;
constexpr float kScale = 1.0f / 11.0f;   // 1 / int(sqrt(128))
constexpr float kEps = 1e-5f;

typedef __bf16 bf16_t;
typedef __attribute__((ext_vector_type(16))) __bf16 v16bf;
typedef __attribute__((ext_vector_type(8)))  __bf16 v8bf;
typedef __attribute__((ext_vector_type(8)))  float  v8f;

// ---------------------------------------------------------------------------
// WMMA fragment helpers (wave32, V_WMMA_F32_16X16X32_BF16; contiguous k order, not the ISA table order).
// With K innermost in memory, each lane's fragment is contiguous:
//   A (16x32 MxK, row-major):  lane lr -> [k = 16*half .. +15]
//   B (32x16 KxN) from N-major-K-contiguous source: lane lr -> [k = 16*half .. +15]
// ---------------------------------------------------------------------------
static __device__ __forceinline__ v16bf frag_a_rowmajor(const bf16_t* p, int ld) {
  int lane = threadIdx.x & 31;
  int half = lane >> 4, lr = lane & 15;
  return *(const v16bf*)(p + lr * ld + 16 * half);
}

static __device__ __forceinline__ v16bf frag_b_nk(const bf16_t* p, int ld) {
  // source: row-major N x K (B[k][n] = p[n*ld + k]); one 32B load per lane
  int lane = threadIdx.x & 31;
  int half = lane >> 4, lr = lane & 15;
  return *(const v16bf*)(p + lr * ld + 16 * half);
}

static __device__ __forceinline__ v8f wmma_bf16(v16bf a, v16bf b, v8f c) {
  return __builtin_amdgcn_wmma_f32_16x16x32_bf16(false, a, false, b, (short)0, c,
                                                 false, false);
}

// ---------------------------------------------------------------------------
// Kernel 0a: fp32 -> bf16 conversion (row-major copy)
// ---------------------------------------------------------------------------
__global__ void cvt_f32_bf16(const float* __restrict__ s, bf16_t* __restrict__ d,
                             int n) {
  int i = blockIdx.x * blockDim.x + threadIdx.x;
  if (i < n) d[i] = (bf16_t)s[i];
}

// ---------------------------------------------------------------------------
// Kernel 0b: fp32 -> bf16 conversion with per-matrix transpose.
// s: [mats][rows][cols]  ->  d: [mats][cols][rows]   (rows = K, cols = N)
// ---------------------------------------------------------------------------
__global__ void cvt_t_f32_bf16(const float* __restrict__ s, bf16_t* __restrict__ d,
                               int rows, int cols, int mats) {
  int i = blockIdx.x * blockDim.x + threadIdx.x;
  int total = mats * rows * cols;
  if (i >= total) return;
  int m = i / (rows * cols);
  int r = (i / cols) % rows;
  int c = i % cols;
  d[(size_t)m * rows * cols + (size_t)c * rows + r] = (bf16_t)s[i];
}

// ---------------------------------------------------------------------------
// Kernel 1: per-(b,h) projection GEMM + bias + non-affine LayerNorm.
// One wave computes a 16x128 output strip for one of {Q,K,V}.
// Q,K stored [B,H,N,D] (row-major, D contiguous); V stored transposed
// [B,H,E,N] so the attention kernel's B fragments are K-contiguous.
// grid = B*H*(N/16)*3, block = 32.
// ---------------------------------------------------------------------------
__global__ void proj_ln_kernel(const bf16_t* __restrict__ xbf,
                               const bf16_t* __restrict__ wqt,
                               const bf16_t* __restrict__ wkt,
                               const bf16_t* __restrict__ wvt,
                               const float* __restrict__ bq,
                               const float* __restrict__ bk,
                               const float* __restrict__ bv,
                               bf16_t* __restrict__ Q, bf16_t* __restrict__ K,
                               bf16_t* __restrict__ Vt) {
  int t = blockIdx.x;
  int which = t % 3;            t /= 3;
  int it    = t % (kN / 16);    t /= (kN / 16);
  int h     = t % kH;
  int b     = t / kH;
  int lane = threadIdx.x;
  int half = lane >> 4, lr = lane & 15;

  // transposed weights: [H][D][E] (n-major, k contiguous)
  const bf16_t* Wt   = (which == 0 ? wqt : which == 1 ? wkt : wvt) + (size_t)h * kE * kD;
  const float*  bias = (which == 0 ? bq : which == 1 ? bk : bv) + h * kD;
  const bf16_t* xs = xbf + ((size_t)b * kN + it * 16) * kE;

  __shared__ alignas(32) float tile[16 * kD];
  __shared__ float mu_s[16], rs_s[16];

  v16bf afr[4];
#pragma unroll
  for (int ks = 0; ks < 4; ++ks) afr[ks] = frag_a_rowmajor(xs + ks * 32, kE);

#pragma unroll
  for (int ct = 0; ct < 8; ++ct) {
    v8f acc = {};
#pragma unroll
    for (int ks = 0; ks < 4; ++ks) {
      v16bf bfr = frag_b_nk(Wt + (size_t)(ct * 16) * kE + ks * 32, kE);
      acc = wmma_bf16(afr[ks], bfr, acc);
    }
    int n = ct * 16 + lr;
    float bn = bias[n];
#pragma unroll
    for (int r = 0; r < 8; ++r) {
      int m = r + 8 * half;
      tile[m * kD + n] = acc[r] + bn;
    }
  }
  __syncthreads();
  if (lane < 16) {
    float s = 0.f;
    for (int e = 0; e < kD; ++e) s += tile[lane * kD + e];
    float mu = s * (1.0f / kD);
    float v = 0.f;
    for (int e = 0; e < kD; ++e) {
      float d0 = tile[lane * kD + e] - mu;
      v += d0 * d0;
    }
    mu_s[lane] = mu;
    rs_s[lane] = rsqrtf(v * (1.0f / kD) + kEps);
  }
  __syncthreads();

  if (which < 2) {
    // Q/K: row-major strip store, 32B vector stores
    bf16_t* out = (which == 0 ? Q : K) + ((size_t)(b * kH + h) * kN + it * 16) * kD;
    int row = lane & 15, seg = lane >> 4;
    float mu = mu_s[row], rs = rs_s[row];
#pragma unroll
    for (int c = 0; c < 64; c += 16) {
      v16bf o;
#pragma unroll
      for (int i = 0; i < 16; ++i) {
        int col = seg * 64 + c + i;
        o[i] = (bf16_t)((tile[row * kD + col] - mu) * rs);
      }
      *(v16bf*)(out + row * kD + seg * 64 + c) = o;
    }
  } else {
    // V: transposed store Vt[b,h,e,n] -- 16 contiguous n per e-row
    bf16_t* outv = Vt + (size_t)(b * kH + h) * kE * kN + it * 16;
    for (int e = lane; e < kE; e += 32) {
      v16bf o;
#pragma unroll
      for (int n = 0; n < 16; ++n)
        o[n] = (bf16_t)((tile[n * kD + e] - mu_s[n]) * rs_s[n]);
      *(v16bf*)(outv + (size_t)e * kN) = o;
    }
  }
}

// ---------------------------------------------------------------------------
// Kernel 2: batch-axis softmax denominators.
// Z[h,n,m] = sum_b exp(Q[b,h,n,:].K[b,h,m,:] / 11)
// One wave per (h, n-tile, m-tile), loops over all 64 batches in registers.
// grid = H*32*32, block = 32.
// ---------------------------------------------------------------------------
__global__ void zsum_kernel(const bf16_t* __restrict__ Q,
                            const bf16_t* __restrict__ K,
                            float* __restrict__ Z) {
  int t = blockIdx.x;
  int jt = t % 32; t /= 32;
  int it = t % 32; t /= 32;
  int h = t;
  int lane = threadIdx.x;
  int half = lane >> 4, lr = lane & 15;

  v8f z = {};
  for (int b = 0; b < kB; ++b) {
    const bf16_t* qp = Q + ((size_t)(b * kH + h) * kN + it * 16) * kD;
    const bf16_t* kp = K + ((size_t)(b * kH + h) * kN + jt * 16) * kD;
    v8f s = {};
#pragma unroll
    for (int ks = 0; ks < 4; ++ks) {
      v16bf a  = frag_a_rowmajor(qp + ks * 32, kD);
      v16bf bb = frag_b_nk(kp + ks * 32, kD);   // B = K^T (K is N x K layout)
      s = wmma_bf16(a, bb, s);
    }
#pragma unroll
    for (int r = 0; r < 8; ++r) z[r] += __expf(s[r] * kScale);
  }
#pragma unroll
  for (int r = 0; r < 8; ++r) {
    int m = r + 8 * half;
    Z[((size_t)h * kN + it * 16 + m) * kN + jt * 16 + lr] = z[r];
  }
}

// ---------------------------------------------------------------------------
// Kernel 3: flash-style attention. Recompute S tiles, P = exp(S/11)/Z,
// bounce P through LDS into A-fragment layout, accumulate P@V (V transposed).
// Output bounced through LDS and written vectorized into A_cat [B,N,H*E].
// grid = B*H*32, block = 32.
// ---------------------------------------------------------------------------
__global__ void attn_pv_kernel(const bf16_t* __restrict__ Q,
                               const bf16_t* __restrict__ K,
                               const bf16_t* __restrict__ Vt,
                               const float* __restrict__ Z,
                               bf16_t* __restrict__ Acat) {
  int t = blockIdx.x;
  int it = t % 32; t /= 32;
  int h  = t % kH;
  int b  = t / kH;
  int lane = threadIdx.x;
  int half = lane >> 4, lr = lane & 15;

  __shared__ alignas(32) bf16_t Pl[16 * 32];
  __shared__ alignas(32) bf16_t Ol[16 * 128];

  const bf16_t* qp = Q + ((size_t)(b * kH + h) * kN + it * 16) * kD;
  v16bf aq[4];
#pragma unroll
  for (int ks = 0; ks < 4; ++ks) aq[ks] = frag_a_rowmajor(qp + ks * 32, kD);

  v8f acc[8];
#pragma unroll
  for (int ct = 0; ct < 8; ++ct)
#pragma unroll
    for (int r = 0; r < 8; ++r) acc[ct][r] = 0.f;

  const bf16_t* vbase = Vt + (size_t)(b * kH + h) * kE * kN;

  for (int mt = 0; mt < 16; ++mt) {
    int m0 = mt * 32;
#pragma unroll
    for (int sub = 0; sub < 2; ++sub) {
      const bf16_t* kp = K + ((size_t)(b * kH + h) * kN + m0 + sub * 16) * kD;
      v8f s = {};
#pragma unroll
      for (int ks = 0; ks < 4; ++ks) {
        v16bf bb = frag_b_nk(kp + ks * 32, kD);
        s = wmma_bf16(aq[ks], bb, s);
      }
#pragma unroll
      for (int r = 0; r < 8; ++r) {
        int m = r + 8 * half;  // query row within tile
        size_t zi = ((size_t)h * kN + it * 16 + m) * kN + m0 + sub * 16 + lr;
        float p = __expf(s[r] * kScale) / Z[zi];
        Pl[m * 32 + sub * 16 + lr] = (bf16_t)p;
      }
    }
    __syncthreads();
    v16bf ap = frag_a_rowmajor(Pl, 32);
#pragma unroll
    for (int ct = 0; ct < 8; ++ct) {
      // B fragment from Vt: rows = e (N dim), k = m contiguous
      v16bf bv2 = frag_b_nk(vbase + (size_t)(ct * 16) * kN + m0, kN);
      acc[ct] = wmma_bf16(ap, bv2, acc[ct]);
    }
    __syncthreads();
  }

  // stage output tile to LDS, then vectorized global stores
#pragma unroll
  for (int ct = 0; ct < 8; ++ct) {
#pragma unroll
    for (int r = 0; r < 8; ++r) {
      int m = r + 8 * half;
      Ol[m * 128 + ct * 16 + lr] = (bf16_t)acc[ct][r];
    }
  }
  __syncthreads();
  int row = lane & 15, seg = lane >> 4;
  bf16_t* ob = Acat + ((size_t)b * kN + it * 16) * (kH * kE) + h * kE;
#pragma unroll
  for (int c = 0; c < 64; c += 16) {
    v16bf o = *(const v16bf*)(Ol + row * 128 + seg * 64 + c);
    *(v16bf*)(ob + (size_t)row * (kH * kE) + seg * 64 + c) = o;
  }
}

// ---------------------------------------------------------------------------
// Kernel 4: MLP (relu(A_cat@W1+b1) @ W2 + b2, relu) + residual + affine LN.
// Weights pre-transposed: W1t [128][512], W2t [128][128].
// grid = B*32, block = 32.
// ---------------------------------------------------------------------------
__global__ void mlp_ln_kernel(const bf16_t* __restrict__ Acat,
                              const bf16_t* __restrict__ W1t,
                              const float* __restrict__ b1,
                              const bf16_t* __restrict__ W2t,
                              const float* __restrict__ b2,
                              const float* __restrict__ x,
                              const float* __restrict__ gamma,
                              const float* __restrict__ beta,
                              float* __restrict__ out) {
  int t = blockIdx.x;
  int it = t % 32;
  int b  = t / 32;
  int lane = threadIdx.x;
  int half = lane >> 4, lr = lane & 15;

  __shared__ alignas(32) bf16_t H1[16 * kE];
  __shared__ alignas(32) float  Y[16 * kE];
  __shared__ float mu_s[16], rs_s[16];

  const bf16_t* ar = Acat + ((size_t)b * kN + it * 16) * (kH * kE);

  // GEMM1: [16 x 512] @ [512 x 128], ks-outer so A fragments load once
  {
    v8f acc[8];
#pragma unroll
    for (int ct = 0; ct < 8; ++ct)
#pragma unroll
      for (int r = 0; r < 8; ++r) acc[ct][r] = 0.f;
    for (int ks = 0; ks < 16; ++ks) {
      v16bf a = frag_a_rowmajor(ar + ks * 32, kH * kE);
#pragma unroll
      for (int ct = 0; ct < 8; ++ct) {
        v16bf bb = frag_b_nk(W1t + (size_t)(ct * 16) * (kH * kE) + ks * 32, kH * kE);
        acc[ct] = wmma_bf16(a, bb, acc[ct]);
      }
    }
#pragma unroll
    for (int ct = 0; ct < 8; ++ct) {
      int n = ct * 16 + lr;
      float bn = b1[n];
#pragma unroll
      for (int r = 0; r < 8; ++r) {
        int m = r + 8 * half;
        float v = acc[ct][r] + bn;
        H1[m * kE + n] = (bf16_t)(v > 0.f ? v : 0.f);
      }
    }
  }
  __syncthreads();

  // GEMM2: [16 x 128] @ [128 x 128]
  {
    v8f acc[8];
#pragma unroll
    for (int ct = 0; ct < 8; ++ct)
#pragma unroll
      for (int r = 0; r < 8; ++r) acc[ct][r] = 0.f;
#pragma unroll
    for (int ks = 0; ks < 4; ++ks) {
      v16bf a = frag_a_rowmajor(H1 + ks * 32, kE);
#pragma unroll
      for (int ct = 0; ct < 8; ++ct) {
        v16bf bb = frag_b_nk(W2t + (size_t)(ct * 16) * kE + ks * 32, kE);
        acc[ct] = wmma_bf16(a, bb, acc[ct]);
      }
    }
#pragma unroll
    for (int ct = 0; ct < 8; ++ct) {
      int n = ct * 16 + lr;
      float bn = b2[n];
#pragma unroll
      for (int r = 0; r < 8; ++r) {
        int m = r + 8 * half;
        float v = acc[ct][r] + bn;
        v = v > 0.f ? v : 0.f;
        Y[m * kE + n] = v + x[((size_t)b * kN + it * 16 + m) * kE + n];
      }
    }
  }
  __syncthreads();

  if (lane < 16) {
    float s = 0.f;
    for (int e = 0; e < kE; ++e) s += Y[lane * kE + e];
    float mu = s * (1.0f / kE);
    float v = 0.f;
    for (int e = 0; e < kE; ++e) {
      float d0 = Y[lane * kE + e] - mu;
      v += d0 * d0;
    }
    mu_s[lane] = mu;
    rs_s[lane] = rsqrtf(v * (1.0f / kE) + kEps);
  }
  __syncthreads();

  int row = lane & 15, seg = lane >> 4;
  float mu = mu_s[row], rs = rs_s[row];
  float* ob = out + ((size_t)b * kN + it * 16 + row) * kE + seg * 64;
#pragma unroll
  for (int c = 0; c < 64; c += 4) {
    int col = seg * 64 + c;
    float4 o;
    o.x = (Y[row * kE + col + 0] - mu) * rs * gamma[col + 0] + beta[col + 0];
    o.y = (Y[row * kE + col + 1] - mu) * rs * gamma[col + 1] + beta[col + 1];
    o.z = (Y[row * kE + col + 2] - mu) * rs * gamma[col + 2] + beta[col + 2];
    o.w = (Y[row * kE + col + 3] - mu) * rs * gamma[col + 3] + beta[col + 3];
    *(float4*)(ob + c) = o;
  }
}

// ---------------------------------------------------------------------------
// Host launcher
// ---------------------------------------------------------------------------
extern "C" void kernel_launch(void* const* d_in, const int* in_sizes, int n_in,
                              void* d_out, int out_size, void* d_ws, size_t ws_size,
                              hipStream_t stream) {
  (void)in_sizes; (void)n_in; (void)out_size; (void)ws_size;

  const float* x_f  = (const float*)d_in[0];
  const float* Wq_f = (const float*)d_in[1];
  const float* bq_f = (const float*)d_in[2];
  const float* Wk_f = (const float*)d_in[3];
  const float* bk_f = (const float*)d_in[4];
  const float* Wv_f = (const float*)d_in[5];
  const float* bv_f = (const float*)d_in[6];
  const float* W1_f = (const float*)d_in[7];
  const float* b1_f = (const float*)d_in[8];
  const float* W2_f = (const float*)d_in[9];
  const float* b2_f = (const float*)d_in[10];
  const float* gamma_f = (const float*)d_in[11];
  const float* beta_f  = (const float*)d_in[12];
  float* out = (float*)d_out;

  // Workspace carving (256B aligned)
  char* ws = (char*)d_ws;
  size_t off = 0;
  auto carve = [&](size_t bytes) -> char* {
    char* p = ws + off;
    off += (bytes + 255) & ~(size_t)255;
    return p;
  };
  const size_t nX  = (size_t)kB * kN * kE;
  const size_t nW  = (size_t)kH * kE * kD;
  const size_t nW1 = (size_t)(kH * kE) * kE;
  const size_t nW2 = (size_t)kE * kE;
  const size_t nQ  = (size_t)kB * kH * kN * kD;
  const size_t nZ  = (size_t)kH * kN * kN;
  const size_t nA  = (size_t)kB * kN * kH * kE;

  bf16_t* xbf  = (bf16_t*)carve(nX * 2);
  bf16_t* wqt  = (bf16_t*)carve(nW * 2);
  bf16_t* wkt  = (bf16_t*)carve(nW * 2);
  bf16_t* wvt  = (bf16_t*)carve(nW * 2);
  bf16_t* w1t  = (bf16_t*)carve(nW1 * 2);
  bf16_t* w2t  = (bf16_t*)carve(nW2 * 2);
  bf16_t* Qb   = (bf16_t*)carve(nQ * 2);
  bf16_t* Kb   = (bf16_t*)carve(nQ * 2);
  bf16_t* Vtb  = (bf16_t*)carve(nQ * 2);
  float*  Zb   = (float*)carve(nZ * 4);
  bf16_t* Ab   = (bf16_t*)carve(nA * 2);

  // 0) precision conversion (+ weight transposes so B fragments are K-contiguous)
  {
    int blocks = (int)((nX + 255) / 256);
    cvt_f32_bf16<<<blocks, 256, 0, stream>>>(x_f, xbf, (int)nX);
  }
  auto cvtT = [&](const float* s, bf16_t* d, int rows, int cols, int mats) {
    int total = rows * cols * mats;
    cvt_t_f32_bf16<<<(total + 255) / 256, 256, 0, stream>>>(s, d, rows, cols, mats);
  };
  cvtT(Wq_f, wqt, kE, kD, kH);        // [H][E][D] -> [H][D][E]
  cvtT(Wk_f, wkt, kE, kD, kH);
  cvtT(Wv_f, wvt, kE, kD, kH);
  cvtT(W1_f, w1t, kH * kE, kE, 1);    // [512][128] -> [128][512]
  cvtT(W2_f, w2t, kE, kE, 1);         // [128][128] -> [128][128]^T

  // 1) QKV projection + LN (V stored transposed)
  proj_ln_kernel<<<kB * kH * (kN / 16) * 3, 32, 0, stream>>>(
      xbf, wqt, wkt, wvt, bq_f, bk_f, bv_f, Qb, Kb, Vtb);

  // 2) batch-softmax denominators
  zsum_kernel<<<kH * 32 * 32, 32, 0, stream>>>(Qb, Kb, Zb);

  // 3) attention (recompute S, P@V) -> A_cat
  attn_pv_kernel<<<kB * kH * 32, 32, 0, stream>>>(Qb, Kb, Vtb, Zb, Ab);

  // 4) MLP + residual + affine LN -> out
  mlp_ln_kernel<<<kB * 32, 32, 0, stream>>>(Ab, w1t, b1_f, w2t, b2_f, x_f,
                                            gamma_f, beta_f, out);
}
